// RelativeMultiHeadAttention_49117245997575
// MI455X (gfx1250) — matrix operand order repaired and checked by reading, not compiled, not run on hardware
//
#include <hip/hip_runtime.h>

// ---------------------------------------------------------------------------
// Types for CDNA5 WMMA (wave32, 16x16x32 bf16 -> f32) and TDM descriptors
// ---------------------------------------------------------------------------
typedef __attribute__((ext_vector_type(16))) __bf16 v16bf;
typedef __attribute__((ext_vector_type(8)))  __bf16 v8bf;
typedef __attribute__((ext_vector_type(8)))  float  v8f;
typedef __attribute__((ext_vector_type(4)))  unsigned int v4u;
typedef __attribute__((ext_vector_type(8)))  int v8i;
typedef __attribute__((ext_vector_type(4)))  int v4i;

typedef __attribute__((address_space(3))) __bf16 lds_bf16_t;
typedef __attribute__((address_space(1))) v4i g_v4i_t;  // global v4i (AS1)
typedef __attribute__((address_space(3))) v4i l_v4i_t;  // LDS v4i (AS3)

union ABf { v16bf v; v8bf h[2]; };

__device__ __forceinline__ float hmax16(float x) {
#pragma unroll
  for (int m = 1; m < 16; m <<= 1) x = fmaxf(x, __shfl_xor(x, m, 16));
  return x;
}
__device__ __forceinline__ float hsum16(float x) {
#pragma unroll
  for (int m = 1; m < 16; m <<= 1) x += __shfl_xor(x, m, 16);
  return x;
}

// Async global->LDS copy of 16 bytes (gfx1250 GLOBAL_LOAD_ASYNC_TO_LDS_B128).
__device__ __forceinline__ void async_ld16B(const void* g, void* l) {
  __builtin_amdgcn_global_load_async_to_lds_b128(
      (g_v4i_t*)(unsigned long long)g, (l_v4i_t*)l, 0, 0);
}
__device__ __forceinline__ void wait_async0() {
  asm volatile("s_wait_asynccnt 0x0" ::: "memory");
}

// ---------------------------------------------------------------------------
// fp32 -> bf16 conversion (grid-stride)
// ---------------------------------------------------------------------------
__global__ void cvt_f32_bf16(const float* __restrict__ src, __bf16* __restrict__ dst, int n) {
  for (int i = blockIdx.x * blockDim.x + threadIdx.x; i < n; i += gridDim.x * blockDim.x)
    dst[i] = (__bf16)src[i];
}

// Ek (129 x 64) -> bf16 padded to (192 x 64) with zeros (so GEMM needs no bounds checks)
__global__ void pad_ek_bf16(const float* __restrict__ Ek, __bf16* __restrict__ dst) {
  int i = blockIdx.x * blockDim.x + threadIdx.x;
  if (i < 192 * 64) {
    int row = i >> 6;
    dst[i] = (row < 129) ? (__bf16)Ek[i] : (__bf16)0.0f;
  }
}

// ---------------------------------------------------------------------------
// Generic GEMM: C[M,N] = A[M,K] @ B[N,K]^T (+bias), bf16 inputs, f32 accum.
// Workgroup = 4 waves; tile 64(M) x 64(N); wave w owns rows [w*16, w*16+16).
// B tile staged in LDS via async-to-LDS loads (ASYNCcnt path), padded stride
// 40 halves to dodge bank conflicts. M,N multiples of 64; K multiple of 32.
// ---------------------------------------------------------------------------
__global__ __launch_bounds__(128) void gemm_xwt_bf16(
    const __bf16* __restrict__ A, int lda,
    const __bf16* __restrict__ B, int ldb,
    const float* __restrict__ bias,
    void* __restrict__ C, int ldc, int out_bf16, int K) {
  __shared__ __align__(16) __bf16 Bs[64 * 40];

  const int tid  = threadIdx.x;
  const int w    = tid >> 5;
  const int lane = tid & 31;
  const int half = lane >> 4;
  const int ln   = lane & 15;
  const size_t m0 = (size_t)blockIdx.y * 64;
  const size_t n0 = (size_t)blockIdx.x * 64;

  v8f acc[4] = {};

  const __bf16* arow = A + (m0 + (size_t)(w * 16 + ln)) * (size_t)lda + half * 8;
  const int srow = tid >> 1, schunk = tid & 1;
  const __bf16* bsrc = B + (n0 + (size_t)srow) * (size_t)ldb + schunk * 16;
  __bf16* bdst = &Bs[srow * 40 + schunk * 16];

  for (int k0 = 0; k0 < K; k0 += 32) {
    __syncthreads();
    async_ld16B(bsrc + k0, bdst);
    async_ld16B(bsrc + k0 + 8, bdst + 8);
    if (k0 + 32 < K) __builtin_prefetch(bsrc + k0 + 32, 0, 1);
    wait_async0();
    __syncthreads();

    ABf a;
    a.h[0] = *(const v8bf*)(arow + k0);
    a.h[1] = *(const v8bf*)(arow + k0 + 16);
#pragma unroll
    for (int nt = 0; nt < 4; ++nt) {
      ABf bf;
      const __bf16* bp = &Bs[(nt * 16 + ln) * 40 + half * 8];
      bf.h[0] = *(const v8bf*)bp;
      bf.h[1] = *(const v8bf*)(bp + 16);
      acc[nt] = __builtin_amdgcn_wmma_f32_16x16x32_bf16(
          false, a.v, false, bf.v, (short)0, acc[nt], false, false);
    }
  }

#pragma unroll
  for (int nt = 0; nt < 4; ++nt) {
    size_t n = n0 + nt * 16 + ln;
    float bv = bias ? bias[n] : 0.0f;
#pragma unroll
    for (int r = 0; r < 8; ++r) {
      size_t m = m0 + w * 16 + r + 8 * half;
      float vv = acc[nt][r] + bv;
      if (out_bf16) ((__bf16*)C)[m * (size_t)ldc + n] = (__bf16)vv;
      else          ((float*)C)[m * (size_t)ldc + n] = vv;
    }
  }
}

// ---------------------------------------------------------------------------
// Fused flash-style relative-position attention.
// Grid: B*H*(L/64) = 1024 workgroups, 128 threads (4 waves).
// K tile is fetched by the Tensor Data Mover (one 2D D# per 32-column block,
// LDS row padding 128B->+16B giving the 72-half stride); V tile is staged
// transposed via VGPRs; P goes through LDS for the C->A layout change.
// ---------------------------------------------------------------------------
__global__ __launch_bounds__(128) void attn_rel_fused(
    const __bf16* __restrict__ qp, const __bf16* __restrict__ kp,
    const __bf16* __restrict__ vp, const float* __restrict__ P2,
    const float* __restrict__ Ev, __bf16* __restrict__ out) {
  __shared__ float  evs[129 * 64];      // Ev staged once (33 KB)
  __shared__ float  wbuf[4][16][132];   // per-wave bucket mass (33 KB)
  __shared__ __align__(16) __bf16 ks[32 * 72];    // K tile (TDM dest, padded)
  __shared__ __align__(16) __bf16 vt[64 * 40];    // V tile transposed
  __shared__ __align__(16) __bf16 ps[4][16 * 40]; // per-wave P staging

  const int tid  = threadIdx.x;
  const int w    = tid >> 5;
  const int lane = tid & 31;
  const int half = lane >> 4;
  const int ln   = lane & 15;
  const int qblk = blockIdx.x & 15;
  const int hh   = (blockIdx.x >> 4) & 15;
  const int b    = blockIdx.x >> 8;
  const int q0   = qblk * 64 + w * 16;  // this wave's first query row

  for (int i = tid; i < 129 * 64; i += 128) evs[i] = Ev[i];
  for (int i = lane; i < 16 * 132; i += 32) (&wbuf[w][0][0])[i] = 0.0f;

  // Q fragments (two K=32 chunks of the 64-dim head)
  const __bf16* qbase =
      qp + ((size_t)(b * 1024 + q0 + ln) * 16 + hh) * 64 + half * 16;
  ABf qa0, qa1;
  qa0.h[0] = *(const v8bf*)(qbase);
  qa0.h[1] = *(const v8bf*)(qbase + 8);
  qa1.h[0] = *(const v8bf*)(qbase + 32);
  qa1.h[1] = *(const v8bf*)(qbase + 40);

  v8f O[4] = {};
  float mrow[8], lrow[8];
#pragma unroll
  for (int r = 0; r < 8; ++r) { mrow[r] = -1e30f; lrow[r] = 0.0f; }

  const int vk = tid & 31, vd0 = (tid >> 5) * 16; // V-transpose staging map
  const unsigned lds_ks = (unsigned)(unsigned long long)(lds_bf16_t*)&ks[0];

  __syncthreads();

  for (int kb0 = 0; kb0 < 1024; kb0 += 32) {
    __syncthreads();
    if (w == 0) { // Tensor Data Mover: 32x64-elem bf16 tile, padded into LDS
      unsigned long long gk = (unsigned long long)(const void*)
          (kp + ((size_t)(b * 1024 + kb0) * 16 + hh) * 64);
      v4u g0;
      g0[0] = 1u;                                   // count=1, load, user D#
      g0[1] = lds_ks;                               // lds_addr (bytes)
      g0[2] = (unsigned)(gk & 0xffffffffu);         // global_addr lo
      g0[3] = (unsigned)((gk >> 32) & 0x01ffffffu)  // global_addr hi (57b)
              | 0x80000000u;                        // type=2 ("image")
      v8i g1;
      g1[0] = (int)((1u << 16)        // data_size = 2 bytes
                  | (1u << 20)        // pad_enable
                  | (4u << 22)        // pad_interval: 32 DWORDs (128 B)
                  | (3u << 25));      // pad_amount: 4 DWORDs (16 B)
      g1[1] = (int)(64u << 16);       // tensor_dim0 = 64 elements
      g1[2] = (int)(32u << 16);       // tensor_dim1 = 32 rows
      g1[3] = (int)(64u << 16);       // tile_dim0 = 64
      g1[4] = 32;                     // tile_dim1 = 32, tile_dim2 = 0
      g1[5] = 1024;                   // tensor_dim0_stride = 1024 elements
      g1[6] = 0;
      g1[7] = 0;                      // tensor_dim1_stride unused (2D)
      v4i gz4 = {0, 0, 0, 0};
      v8i gz8 = {0, 0, 0, 0, 0, 0, 0, 0};
      __builtin_amdgcn_tensor_load_to_lds(g0, g1, gz4, gz4, gz8, 0);
      __builtin_amdgcn_s_wait_tensorcnt((short)0);
    }
    { // stage V transposed: vt[d][k]
      const __bf16* vsrc =
          vp + ((size_t)(b * 1024 + kb0 + vk) * 16 + hh) * 64 + vd0;
      v8bf x0 = *(const v8bf*)vsrc;
      v8bf x1 = *(const v8bf*)(vsrc + 8);
#pragma unroll
      for (int j = 0; j < 8; ++j) vt[(vd0 + j) * 40 + vk] = x0[j];
#pragma unroll
      for (int j = 0; j < 8; ++j) vt[(vd0 + 8 + j) * 40 + vk] = x1[j];
    }
    __syncthreads();

    // S = Q K^T : two 16x16 tiles (k-local 0..15 and 16..31)
    v8f Sa = {}, Sb = {};
    {
      ABf bk;
      const __bf16* bp = &ks[ln * 72 + half * 16];
      bk.h[0] = *(const v8bf*)bp; bk.h[1] = *(const v8bf*)(bp + 8);
      Sa = __builtin_amdgcn_wmma_f32_16x16x32_bf16(false, qa0.v, false, bk.v, (short)0, Sa, false, false);
      bp += 32;
      bk.h[0] = *(const v8bf*)bp; bk.h[1] = *(const v8bf*)(bp + 8);
      Sa = __builtin_amdgcn_wmma_f32_16x16x32_bf16(false, qa1.v, false, bk.v, (short)0, Sa, false, false);
      bp = &ks[(16 + ln) * 72 + half * 16];
      bk.h[0] = *(const v8bf*)bp; bk.h[1] = *(const v8bf*)(bp + 8);
      Sb = __builtin_amdgcn_wmma_f32_16x16x32_bf16(false, qa0.v, false, bk.v, (short)0, Sb, false, false);
      bp += 32;
      bk.h[0] = *(const v8bf*)bp; bk.h[1] = *(const v8bf*)(bp + 8);
      Sb = __builtin_amdgcn_wmma_f32_16x16x32_bf16(false, qa1.v, false, bk.v, (short)0, Sb, false, false);
    }

    // relative scores + scale + online softmax + bucket mass + P staging
#pragma unroll
    for (int r = 0; r < 8; ++r) {
      int M  = r + 8 * half;
      int qg = q0 + M;
      const float* p2row = P2 + ((size_t)(b * 1024 + qg) * 16 + hh) * 192;
      int ka = kb0 + ln, kb = kb0 + 16 + ln;
      int ra = min(max(ka - qg, -64), 64) + 64;
      int rb = min(max(kb - qg, -64), 64) + 64;
      float sa = (Sa[r] + p2row[ra]) * 0.125f;   // 1/sqrt(64)
      float sb = (Sb[r] + p2row[rb]) * 0.125f;
      float rm   = hmax16(fmaxf(sa, sb));
      float mnew = fmaxf(mrow[r], rm);
      float alpha = __expf(mrow[r] - mnew);
      mrow[r] = mnew;
      float pa = __expf(sa - mnew);
      float pb = __expf(sb - mnew);
      lrow[r] = lrow[r] * alpha + hsum16(pa + pb);
#pragma unroll
      for (int nt = 0; nt < 4; ++nt) O[nt][r] *= alpha;
      if (alpha != 1.0f) {
        for (int t = ln; t < 129; t += 16) wbuf[w][M][t] *= alpha;
      }
      atomicAdd(&wbuf[w][M][ra], pa);
      atomicAdd(&wbuf[w][M][rb], pb);
      ps[w][M * 40 + ln]      = (__bf16)pa;
      ps[w][M * 40 + 16 + ln] = (__bf16)pb;
    }
    __syncthreads();

    // O += P V  (P: 16x32 A-frag from LDS transpose; V: 32x64 via 4 B-frags)
    ABf pfr;
    {
      const __bf16* pp = &ps[w][ln * 40 + half * 8];
      pfr.h[0] = *(const v8bf*)pp;
      pfr.h[1] = *(const v8bf*)(pp + 16);
    }
#pragma unroll
    for (int nt = 0; nt < 4; ++nt) {
      ABf vf;
      const __bf16* vpp = &vt[(nt * 16 + ln) * 40 + half * 8];
      vf.h[0] = *(const v8bf*)vpp;
      vf.h[1] = *(const v8bf*)(vpp + 16);
      O[nt] = __builtin_amdgcn_wmma_f32_16x16x32_bf16(
          false, pfr.v, false, vf.v, (short)0, O[nt], false, false);
    }
  }

  // Epilogue: out = (O + sum_r w[q,r]*Ev[r]) / l
#pragma unroll
  for (int r = 0; r < 8; ++r) {
    int M = r + 8 * half;
    float rel[4] = {0.f, 0.f, 0.f, 0.f};
    for (int rb = 0; rb < 129; ++rb) {
      float wv = wbuf[w][M][rb];
#pragma unroll
      for (int nt = 0; nt < 4; ++nt)
        rel[nt] += wv * evs[rb * 64 + nt * 16 + ln];
    }
    float inv = 1.0f / lrow[r];
    int qg = q0 + M;
    __bf16* orow = out + ((size_t)(b * 1024 + qg) * 16 + hh) * 64;
#pragma unroll
    for (int nt = 0; nt < 4; ++nt)
      orow[nt * 16 + ln] = (__bf16)((O[nt][r] + rel[nt]) * inv);
  }
}

// ---------------------------------------------------------------------------
// Host-side orchestration
// ---------------------------------------------------------------------------
extern "C" void kernel_launch(void* const* d_in, const int* in_sizes, int n_in,
                              void* d_out, int out_size, void* d_ws, size_t ws_size,
                              hipStream_t stream) {
  const float* q  = (const float*)d_in[0];
  const float* k  = (const float*)d_in[1];
  const float* v  = (const float*)d_in[2];
  const float* Wq = (const float*)d_in[3];
  const float* bq = (const float*)d_in[4];
  const float* Wk = (const float*)d_in[5];
  const float* bk = (const float*)d_in[6];
  const float* Wv = (const float*)d_in[7];
  const float* bv = (const float*)d_in[8];
  const float* Wo = (const float*)d_in[9];
  const float* bo = (const float*)d_in[10];
  const float* Ek = (const float*)d_in[11];
  const float* Ev = (const float*)d_in[12];

  const size_t NBLD = 4ull * 1024 * 1024;  // B*L*D elements
  const size_t NDD  = 1024ull * 1024;      // D*D elements

  char* ws = (char*)d_ws;
  size_t off = 0;
  auto take = [&](size_t bytes) -> char* {
    char* p = ws + off;
    off = (off + bytes + 255) & ~(size_t)255;
    return p;
  };

  __bf16* qb  = (__bf16*)take(NBLD * 2);
  __bf16* kb_ = (__bf16*)take(NBLD * 2);
  __bf16* vb_ = (__bf16*)take(NBLD * 2);
  __bf16* Wqb = (__bf16*)take(NDD * 2);
  __bf16* Wkb = (__bf16*)take(NDD * 2);
  __bf16* Wvb = (__bf16*)take(NDD * 2);
  __bf16* Wob = (__bf16*)take(NDD * 2);
  __bf16* ekp = (__bf16*)take(192 * 64 * 2);
  __bf16* qpb = (__bf16*)take(NBLD * 2);
  __bf16* kpb = (__bf16*)take(NBLD * 2);
  __bf16* vpb = (__bf16*)take(NBLD * 2);
  float*  P2  = (float*)take(65536ull * 192 * 4);
  __bf16* aob = (__bf16*)take(NBLD * 2);

  // fp32 -> bf16 conversions
  cvt_f32_bf16<<<2048, 256, 0, stream>>>(q,  qb,  (int)NBLD);
  cvt_f32_bf16<<<2048, 256, 0, stream>>>(k,  kb_, (int)NBLD);
  cvt_f32_bf16<<<2048, 256, 0, stream>>>(v,  vb_, (int)NBLD);
  cvt_f32_bf16<<<1024, 256, 0, stream>>>(Wq, Wqb, (int)NDD);
  cvt_f32_bf16<<<1024, 256, 0, stream>>>(Wk, Wkb, (int)NDD);
  cvt_f32_bf16<<<1024, 256, 0, stream>>>(Wv, Wvb, (int)NDD);
  cvt_f32_bf16<<<1024, 256, 0, stream>>>(Wo, Wob, (int)NDD);
  pad_ek_bf16<<<(192 * 64 + 255) / 256, 256, 0, stream>>>(Ek, ekp);

  // Q/K/V projections: (4096 x 1024) = X @ W^T + b   -> bf16
  dim3 gP(16, 64);
  gemm_xwt_bf16<<<gP, 128, 0, stream>>>(qb,  1024, Wqb, 1024, bq, qpb, 1024, 1, 1024);
  gemm_xwt_bf16<<<gP, 128, 0, stream>>>(kb_, 1024, Wkb, 1024, bk, kpb, 1024, 1, 1024);
  gemm_xwt_bf16<<<gP, 128, 0, stream>>>(vb_, 1024, Wvb, 1024, bv, vpb, 1024, 1, 1024);

  // P2[(b,q,h), r] = qp . Ek[r] : (65536 x 64) @ (192 x 64)^T -> f32 (ldc 192)
  dim3 gR(3, 1024);
  gemm_xwt_bf16<<<gR, 128, 0, stream>>>(qpb, 64, ekp, 64, nullptr, P2, 192, 0, 64);

  // Fused attention: 1024 workgroups (B*H*L/64), 4 waves each
  attn_rel_fused<<<1024, 128, 0, stream>>>(qpb, kpb, vpb, P2, Ev, aob);

  // Output projection -> f32 d_out
  gemm_xwt_bf16<<<gP, 128, 0, stream>>>(aob, 1024, Wob, 1024, bo, (float*)d_out, 1024, 0, 1024);
}
